// Retention_67602785239565
// MI455X (gfx1250) — matrix operand order repaired and checked by reading, not compiled, not run on hardware
//
#include <hip/hip_runtime.h>

typedef __bf16 bf16;
typedef __attribute__((ext_vector_type(16))) __bf16 v16bf;
typedef __attribute__((ext_vector_type(8)))  __bf16 v8bf;
typedef __attribute__((ext_vector_type(8)))  float  v8f;

#define BATCH 16
#define SEQ   1024
#define DIM   768
#define N3    2304

// log2(0.99), 1/sqrt(768)
#define LOG2A   (-0.014499570f)
#define RSQRT_D (0.03608439182f)

__device__ __forceinline__ v8f wmma_bf(v16bf a, v16bf b, v8f c) {
  // D = A(16x32 bf16) * B(32x16 bf16) + C(16x16 f32)
  return __builtin_amdgcn_wmma_f32_16x16x32_bf16(false, a, false, b, (short)0, c,
                                                 false, false);
}

__device__ __forceinline__ v16bf cat16(v8bf c0, v8bf c1) {
  return __builtin_shufflevector(c0, c1, 0, 1, 2, 3, 4, 5, 6, 7,
                                 8, 9, 10, 11, 12, 13, 14, 15);
}

// A-matrix fragment (16x32 bf16) from a row-major [M][K] source.
// Lane L (L<16): M=L, K = {0..7, 16..23}; lane L>=16: M=L-16, K = {8..15, 24..31}.
// Two contiguous 16-byte chunks per lane.
__device__ __forceinline__ v16bf load_a_frag(const bf16* __restrict__ base, int ld) {
  const int lane = threadIdx.x & 31;
  const bf16* p = base + (size_t)(lane & 15) * ld + ((lane >> 4) << 3);
  v8bf c0 = *(const v8bf*)(p);
  v8bf c1 = *(const v8bf*)(p + 16);
  return cat16(c0, c1);
}

// B-matrix fragment (32x16 bf16, B[k][n]) from a source stored [N][K] row-major.
// Lane L (L<16): N=L, K = {0..7, 16..23}; lane L>=16: N=L-16, K = {8..15, 24..31}.
// Two contiguous 16-byte chunks per lane.
__device__ __forceinline__ v16bf load_b_frag(const bf16* __restrict__ base, int ld) {
  const int lane = threadIdx.x & 31;
  const bf16* p = base + (size_t)(lane & 15) * ld + ((lane >> 4) << 3);
  return cat16(*(const v8bf*)(p), *(const v8bf*)(p + 16));
}

__device__ __forceinline__ void splitf(float x, bf16& h, bf16& l) {
  bf16 hh = (bf16)x;
  h = hh;
  l = (bf16)(x - (float)hh);
}

// ---------------------------------------------------------------------------
// Kernel 1: split x (f32) into bf16 hi/lo, same [B*S][D] row-major layout.
// ---------------------------------------------------------------------------
__global__ __launch_bounds__(256) void split_x_kernel(
    const float* __restrict__ x, bf16* __restrict__ Xh, bf16* __restrict__ Xl) {
  const size_t i = (size_t)blockIdx.x * 256 + threadIdx.x;
  bf16 h, l;
  splitf(x[i], h, l);
  Xh[i] = h;
  Xl[i] = l;
}

// ---------------------------------------------------------------------------
// Kernel 2: split + transpose W_qkv [K=768][N=2304] -> WT [N][K] bf16 hi/lo.
// ---------------------------------------------------------------------------
__global__ __launch_bounds__(256) void split_wT_kernel(
    const float* __restrict__ W, bf16* __restrict__ WTh, bf16* __restrict__ WTl) {
  const size_t i = (size_t)blockIdx.x * 256 + threadIdx.x;
  const int n = (int)(i / DIM);
  const int k = (int)(i % DIM);
  bf16 h, l;
  splitf(W[(size_t)k * N3 + n], h, l);
  WTh[i] = h;
  WTl[i] = l;
}

// ---------------------------------------------------------------------------
// Kernel 3: QKV projection GEMM (bf16x3), fused bias + decay scaling + split.
//   qkv[m,n] = sum_k x[m,k] W[k,n] + b[n]
//   Q' = q * alpha^s / sqrt(D)    -> Qh/Ql  [b][s][d] row-major
//   K' = k * alpha^(-s)           -> Kh/Kl  [b][s][d] row-major
//   V                             -> Vth/Vtl [b][d][s]  (transposed)
// Block = 256 threads = 8 waves; tile 128(M) x 64(N); wave tile 32x32.
// ---------------------------------------------------------------------------
__global__ __launch_bounds__(256) void qkv_gemm_kernel(
    const bf16* __restrict__ Xh, const bf16* __restrict__ Xl,
    const bf16* __restrict__ WTh, const bf16* __restrict__ WTl,
    const float* __restrict__ bias,
    bf16* __restrict__ Qh, bf16* __restrict__ Ql,
    bf16* __restrict__ Kh, bf16* __restrict__ Kl,
    bf16* __restrict__ Vth, bf16* __restrict__ Vtl) {
  const int nblk = blockIdx.x * 64;
  const int mblk = blockIdx.y * 128;
  const int w = threadIdx.x >> 5;
  const int lane = threadIdx.x & 31;
  const int m0 = mblk + (w & 3) * 32;
  const int n0 = nblk + (w >> 2) * 32;

  v8f acc00 = {}, acc01 = {}, acc10 = {}, acc11 = {};

  const bf16* xh0 = Xh + (size_t)m0 * DIM;
  const bf16* xl0 = Xl + (size_t)m0 * DIM;
  const bf16* wh0 = WTh + (size_t)n0 * DIM;
  const bf16* wl0 = WTl + (size_t)n0 * DIM;

  for (int kk = 0; kk < DIM; kk += 32) {
    __builtin_prefetch(xh0 + kk + 64, 0, 1);
    __builtin_prefetch(wh0 + kk + 64, 0, 1);
    v16bf a0h = load_a_frag(xh0 + kk, DIM);
    v16bf a0l = load_a_frag(xl0 + kk, DIM);
    v16bf a1h = load_a_frag(xh0 + (size_t)16 * DIM + kk, DIM);
    v16bf a1l = load_a_frag(xl0 + (size_t)16 * DIM + kk, DIM);
    {
      v16bf bh = load_b_frag(wh0 + kk, DIM);
      v16bf bl = load_b_frag(wl0 + kk, DIM);
      acc00 = wmma_bf(a0h, bh, acc00);
      acc00 = wmma_bf(a0h, bl, acc00);
      acc00 = wmma_bf(a0l, bh, acc00);
      acc10 = wmma_bf(a1h, bh, acc10);
      acc10 = wmma_bf(a1h, bl, acc10);
      acc10 = wmma_bf(a1l, bh, acc10);
    }
    {
      v16bf bh = load_b_frag(wh0 + (size_t)16 * DIM + kk, DIM);
      v16bf bl = load_b_frag(wl0 + (size_t)16 * DIM + kk, DIM);
      acc01 = wmma_bf(a0h, bh, acc01);
      acc01 = wmma_bf(a0h, bl, acc01);
      acc01 = wmma_bf(a0l, bh, acc01);
      acc11 = wmma_bf(a1h, bh, acc11);
      acc11 = wmma_bf(a1h, bl, acc11);
      acc11 = wmma_bf(a1l, bh, acc11);
    }
  }

  v8f accs[2][2] = {{acc00, acc01}, {acc10, acc11}};

#pragma unroll
  for (int i = 0; i < 2; ++i) {
    const int mbase = m0 + i * 16 + ((lane >> 4) << 3);  // 8 consecutive rows
    const int bidx = mbase >> 10;
    const int srow = mbase & (SEQ - 1);
#pragma unroll
    for (int j = 0; j < 2; ++j) {
      const int n = n0 + j * 16 + (lane & 15);
      const float bv = bias[n];
      const int seg = n / DIM;  // uniform within subtile (768 % 16 == 0)
      const int d = n - seg * DIM;
      if (seg == 0) {  // Q: scale alpha^s / sqrt(D)
        bf16* qh = Qh + (size_t)(bidx * SEQ + srow) * DIM + d;
        bf16* ql = Ql + (size_t)(bidx * SEQ + srow) * DIM + d;
#pragma unroll
        for (int r = 0; r < 8; ++r) {
          float sc = exp2f((float)(srow + r) * LOG2A) * RSQRT_D;
          bf16 h, l;
          splitf((accs[i][j][r] + bv) * sc, h, l);
          qh[(size_t)r * DIM] = h;
          ql[(size_t)r * DIM] = l;
        }
      } else if (seg == 1) {  // K: scale alpha^(-t)
        bf16* kh = Kh + (size_t)(bidx * SEQ + srow) * DIM + d;
        bf16* kl = Kl + (size_t)(bidx * SEQ + srow) * DIM + d;
#pragma unroll
        for (int r = 0; r < 8; ++r) {
          float sc = exp2f(-(float)(srow + r) * LOG2A);
          bf16 h, l;
          splitf((accs[i][j][r] + bv) * sc, h, l);
          kh[(size_t)r * DIM] = h;
          kl[(size_t)r * DIM] = l;
        }
      } else {  // V: transposed store [b][d][s], 8 contiguous s per lane
        v8bf vh, vl;
#pragma unroll
        for (int r = 0; r < 8; ++r) {
          bf16 h, l;
          splitf(accs[i][j][r] + bv, h, l);
          vh[r] = h;
          vl[r] = l;
        }
        const size_t off = ((size_t)bidx * DIM + d) * SEQ + srow;
        *(v8bf*)(Vth + off) = vh;
        *(v8bf*)(Vtl + off) = vl;
      }
    }
  }
}

// ---------------------------------------------------------------------------
// Kernel 4: flash-style causal retention.
//   scores = Q'K'^T (decay pre-folded), causal mask, out = scores @ V.
// Block = 256 threads = 8 waves, 32 q-rows per block, t-tiles of 64.
// Score tile 32x64 split as 2x4 16x16 subtiles, one per wave; scores
// round-trip through LDS (bf16 hi/lo) to switch D-layout -> A-layout.
// Each wave then owns a 96-column d-slice of the 32x768 output.
// ---------------------------------------------------------------------------
__global__ __launch_bounds__(256) void retention_kernel(
    const bf16* __restrict__ Qh, const bf16* __restrict__ Ql,
    const bf16* __restrict__ Kh, const bf16* __restrict__ Kl,
    const bf16* __restrict__ Vth, const bf16* __restrict__ Vtl,
    float* __restrict__ out) {
  const int b = blockIdx.y;
  const int sblk = blockIdx.x * 32;
  const int w = threadIdx.x >> 5;
  const int lane = threadIdx.x & 31;

  __shared__ __align__(16) bf16 Sh[32][72];  // +8 pad: conflict-free b128 rows
  __shared__ __align__(16) bf16 Sl[32][72];

  const int ms = w & 1;   // score subtile row index for this wave
  const int ts = w >> 1;  // score subtile col index for this wave
  const int d0 = w * 96;  // this wave's output d-slice

  v8f acc[2][6] = {};

  const int srow = sblk + ms * 16;
  const bf16* qrh = Qh + (size_t)(b * SEQ + srow) * DIM;
  const bf16* qrl = Ql + (size_t)(b * SEQ + srow) * DIM;

  const int ntt = (sblk + 31) / 64 + 1;  // causal: skip tiles above diagonal
  for (int tt = 0; tt < ntt; ++tt) {
    const int tbase = tt * 64;
    // ---- score subtile (ms, ts): Q'[16,768] x K'[16,768]^T, bf16x3 ----
    {
      const int trow = tbase + ts * 16;
      const bf16* krh = Kh + (size_t)(b * SEQ + trow) * DIM;
      const bf16* krl = Kl + (size_t)(b * SEQ + trow) * DIM;
      v8f sc = {};
      for (int kk = 0; kk < DIM; kk += 32) {
        __builtin_prefetch(krh + kk + 64, 0, 1);
        v16bf ah = load_a_frag(qrh + kk, DIM);
        v16bf al = load_a_frag(qrl + kk, DIM);
        v16bf bh = load_b_frag(krh + kk, DIM);
        v16bf bl = load_b_frag(krl + kk, DIM);
        sc = wmma_bf(ah, bh, sc);
        sc = wmma_bf(ah, bl, sc);
        sc = wmma_bf(al, bh, sc);
      }
      const int tg = tbase + ts * 16 + (lane & 15);  // global t of this lane
      const int tl_ = ts * 16 + (lane & 15);         // local col in LDS tile
      const int mb = ms * 16 + ((lane >> 4) << 3);   // local row base
      const int sg = sblk + mb;                      // global s row base
#pragma unroll
      for (int r = 0; r < 8; ++r) {
        float v = (tg <= sg + r) ? sc[r] : 0.0f;  // causal mask (decay folded)
        bf16 h, l;
        splitf(v, h, l);
        Sh[mb + r][tl_] = h;
        Sl[mb + r][tl_] = l;
      }
    }
    __syncthreads();
    // ---- out accumulation: acc[32 x 96] += S[32 x 64] @ V[64 x 96] ----
    {
      const int arow = lane & 15;
      const int acol = (lane >> 4) << 3;
#pragma unroll
      for (int ks = 0; ks < 2; ++ks) {
        v16bf a[2][2];  // [m-subtile][hi/lo] A-fragments from LDS
#pragma unroll
        for (int i = 0; i < 2; ++i) {
          const bf16* p0 = &Sh[i * 16 + arow][ks * 32 + acol];
          const bf16* p1 = &Sl[i * 16 + arow][ks * 32 + acol];
          a[i][0] = cat16(*(const v8bf*)p0, *(const v8bf*)(p0 + 16));
          a[i][1] = cat16(*(const v8bf*)p1, *(const v8bf*)(p1 + 16));
        }
#pragma unroll
        for (int j = 0; j < 6; ++j) {
          const size_t vo =
              ((size_t)b * DIM + d0 + j * 16) * SEQ + tbase + ks * 32;
          v16bf bh = load_b_frag(Vth + vo, SEQ);
          v16bf bl = load_b_frag(Vtl + vo, SEQ);
#pragma unroll
          for (int i = 0; i < 2; ++i) {
            acc[i][j] = wmma_bf(a[i][0], bh, acc[i][j]);
            acc[i][j] = wmma_bf(a[i][0], bl, acc[i][j]);
            acc[i][j] = wmma_bf(a[i][1], bh, acc[i][j]);
          }
        }
      }
    }
    __syncthreads();
  }

  // ---- write output f32 [b][s][d] ----
#pragma unroll
  for (int i = 0; i < 2; ++i) {
    const int sb = sblk + i * 16 + ((lane >> 4) << 3);
#pragma unroll
    for (int j = 0; j < 6; ++j) {
      const int d = d0 + j * 16 + (lane & 15);
      float* op = out + (size_t)(b * SEQ + sb) * DIM + d;
#pragma unroll
      for (int r = 0; r < 8; ++r) op[(size_t)r * DIM] = acc[i][j][r];
    }
  }
}

// ---------------------------------------------------------------------------
// Launch
// ---------------------------------------------------------------------------
extern "C" void kernel_launch(void* const* d_in, const int* in_sizes, int n_in,
                              void* d_out, int out_size, void* d_ws,
                              size_t ws_size, hipStream_t stream) {
  (void)in_sizes; (void)n_in; (void)out_size; (void)ws_size;
  const float* x    = (const float*)d_in[0];
  const float* Wq   = (const float*)d_in[1];
  const float* bias = (const float*)d_in[2];
  float* out = (float*)d_out;

  const size_t EL = (size_t)BATCH * SEQ * DIM;  // 12,582,912 elems
  const size_t WN = (size_t)N3 * DIM;           //  1,769,472 elems

  bf16* p   = (bf16*)d_ws;  // total ~208 MB of workspace
  bf16* Xh  = p;
  bf16* Xl  = p + EL;
  bf16* Qh  = p + 2 * EL;
  bf16* Ql  = p + 3 * EL;
  bf16* Kh  = p + 4 * EL;
  bf16* Kl  = p + 5 * EL;
  bf16* Vth = p + 6 * EL;
  bf16* Vtl = p + 7 * EL;
  bf16* WTh = p + 8 * EL;
  bf16* WTl = p + 8 * EL + WN;

  split_x_kernel<<<dim3((unsigned)(EL / 256)), 256, 0, stream>>>(x, Xh, Xl);
  split_wT_kernel<<<dim3((unsigned)(WN / 256)), 256, 0, stream>>>(Wq, WTh, WTl);
  qkv_gemm_kernel<<<dim3(N3 / 64, (BATCH * SEQ) / 128), 256, 0, stream>>>(
      Xh, Xl, WTh, WTl, bias, Qh, Ql, Kh, Kl, Vth, Vtl);
  retention_kernel<<<dim3(SEQ / 32, BATCH), 256, 0, stream>>>(
      Qh, Ql, Kh, Kl, Vth, Vtl, out);
}
